// BiLSTM_CRF_50680614092981
// MI455X (gfx1250) — matrix operand order repaired and checked by reading, not compiled, not run on hardware
//
#include <hip/hip_runtime.h>

typedef __attribute__((ext_vector_type(16))) __bf16 v16bf;
typedef __attribute__((ext_vector_type(8)))  __bf16 v8bf;
typedef __attribute__((ext_vector_type(4)))  __bf16 v4bf;
typedef __attribute__((ext_vector_type(8)))  float  v8f;

#define B_   32
#define T_   256
#define EMB_ 256
#define HH_  256
#define G4_  1024
#define NTG_ 12
#define NEGV -10000.0f

__device__ __forceinline__ float sigf(float x) { return 1.0f / (1.0f + __expf(-x)); }

__device__ __forceinline__ v16bf cat8(v8bf lo, v8bf hi) {
    return __builtin_shufflevector(lo, hi, 0,1,2,3,4,5,6,7,8,9,10,11,12,13,14,15);
}

// ---------------------------------------------------------------------------
// Pack W_ih_f, W_ih_b, W_hh_f, W_hh_b (each [1024 x 256] f32) into bf16
// B-fragment layout: frag(kt,nt,lane,e) = W[nt*16+lane%16][kt*32+(lane/16)*8+(e%8)+(e/8)*16]
// Linear index inside a matrix: ((kt*64+nt)*32+lane)*16+e  (262144 elems/mat)
// ---------------------------------------------------------------------------
__global__ __launch_bounds__(256) void pack_weights(
    const float* __restrict__ Wihf, const float* __restrict__ Wihb,
    const float* __restrict__ Whhf, const float* __restrict__ Whhb,
    __bf16* __restrict__ pack) {
    int idx = blockIdx.x * 256 + threadIdx.x;      // 4 * 262144 total
    int mat = idx >> 18;
    int r   = idx & 262143;
    int e    = r & 15;
    int lane = (r >> 4) & 31;
    int nt   = (r >> 9) & 63;
    int kt   = r >> 15;
    int n = nt * 16 + (lane & 15);
    int k = kt * 32 + (lane >> 4) * 8 + (e & 7) + (e >> 3) * 16;
    const float* W = (mat == 0) ? Wihf : (mat == 1) ? Wihb : (mat == 2) ? Whhf : Whhb;
    pack[idx] = (__bf16)W[n * 256 + k];
}

// ---------------------------------------------------------------------------
// Embedding gather + one-time f32->bf16 convert:
// xsbf[row=t*32+b][k] = (bf16) embed[sentence[b][t]][k]    (8192 x 256)
// Row-major bf16 is directly A-fragment loadable (two 16B loads per lane).
// ---------------------------------------------------------------------------
__global__ __launch_bounds__(256) void embed_pack(
    const int* __restrict__ sent, const float* __restrict__ embed,
    __bf16* __restrict__ xsbf) {
    int idx = blockIdx.x * 256 + threadIdx.x;   // 524288 total, 4 elems each
    int row = idx >> 6;
    int kq  = (idx & 63) * 4;
    int t = row >> 5;
    int b = row & 31;
    int token = sent[b * T_ + t];
    float4 f = *(const float4*)(embed + (size_t)token * EMB_ + kq);
    v4bf o;
    o[0] = (__bf16)f.x; o[1] = (__bf16)f.y; o[2] = (__bf16)f.z; o[3] = (__bf16)f.w;
    *(v4bf*)(xsbf + (size_t)row * EMB_ + kq) = o;
}

// ---------------------------------------------------------------------------
// Input GEMM (time-parallel): Zx[dir][row][n] = xs @ W_ih^T + bias
// Each wave: 1 M-tile x 4 N-tiles, K=256 in 8 bf16 WMMA steps.
// A fragment loaded once per K-step and reused by 4 WMMAs.
// ---------------------------------------------------------------------------
__global__ __launch_bounds__(256) void input_gemm(
    const __bf16* __restrict__ xsbf, const __bf16* __restrict__ packIH,
    const float* __restrict__ bias_f, const float* __restrict__ bias_b,
    float* __restrict__ Zx) {
    int lane = threadIdx.x & 31;
    int wave = threadIdx.x >> 5;
    int task = blockIdx.x * 8 + wave;          // 16384 tasks: dir x 512 mt x 16 ngroups
    int dir  = task >> 13;
    int rem  = task & 8191;
    int mt   = rem >> 4;                       // 0..511
    int ng   = rem & 15;                       // group of 4 N-tiles
    int q    = lane & 15;
    int hl   = lane >> 4;

    const float* bias = dir ? bias_b : bias_f;
    v8f acc[4];
#pragma unroll
    for (int nl = 0; nl < 4; ++nl) {
        float bv = bias[(ng * 4 + nl) * 16 + q];
#pragma unroll
        for (int r = 0; r < 8; ++r) acc[nl][r] = bv;
    }

    const __bf16* arow = xsbf + (size_t)(mt * 16 + q) * EMB_ + hl * 8;
    const v16bf*  bp0  = (const v16bf*)packIH + (size_t)dir * 16384 + lane;

#pragma unroll
    for (int kt = 0; kt < 8; ++kt) {
        v8bf lo = *(const v8bf*)(arow + kt * 32);
        v8bf hi = *(const v8bf*)(arow + kt * 32 + 16);
        v16bf a = cat8(lo, hi);
#pragma unroll
        for (int nl = 0; nl < 4; ++nl) {
            v16bf bw = bp0[(size_t)kt * 2048 + (ng * 4 + nl) * 32];
            acc[nl] = __builtin_amdgcn_wmma_f32_16x16x32_bf16(
                false, a, false, bw, (short)0, acc[nl], false, false);
        }
    }

#pragma unroll
    for (int nl = 0; nl < 4; ++nl) {
        size_t obase = ((size_t)dir * 8192 + (size_t)mt * 16) * G4_ + (ng * 4 + nl) * 16 + q;
#pragma unroll
        for (int r = 0; r < 8; ++r)
            Zx[obase + (size_t)(r + hl * 8) * G4_] = acc[nl][r];
    }
}

// ---------------------------------------------------------------------------
// Persistent recurrent LSTM: one block per direction, 1024 threads = 32 waves.
// Per step: z = Zx[t] + h @ W_hh^T via WMMA (each wave: 2 M-tiles x 2 N-tiles),
// then gate nonlinearities; c lives in registers, h in LDS (bf16).
// Next step's Zx slice is prefetched (global_prefetch_b8) during current step.
// ---------------------------------------------------------------------------
__global__ __launch_bounds__(1024) void lstm_rec(
    const float* __restrict__ Zx, const __bf16* __restrict__ pack,
    const float* __restrict__ h0, const float* __restrict__ c0,
    float* __restrict__ hs) {
    __shared__ __bf16 hbuf[B_ * HH_];    // 16 KB, A-side activations
    __shared__ __bf16 zbuf[B_ * G4_];    // 64 KB, gate pre-activations

    int dir  = blockIdx.x;
    int tid  = threadIdx.x;
    int lane = tid & 31;
    int wave = tid >> 5;
    int q    = lane & 15;
    int hl   = lane >> 4;

    // element-wise phase mapping: 8 cells per thread
    int eb = tid >> 5;           // batch index 0..31
    int ej = (tid & 31) * 8;     // hidden-slot base

    float creg[8];
#pragma unroll
    for (int u = 0; u < 8; ++u) {
        creg[u] = c0[(size_t)dir * 8192 + eb * HH_ + ej + u];
        hbuf[eb * HH_ + ej + u] = (__bf16)h0[(size_t)dir * 8192 + eb * HH_ + ej + u];
    }
    __syncthreads();

    const __bf16* whh = pack + (size_t)(2 + dir) * 262144;
    const v16bf*  wfrag = (const v16bf*)whh;
    const float*  zdir = Zx + (size_t)dir * 8192 * G4_;

    for (int step = 0; step < T_; ++step) {
        int t = dir ? (T_ - 1 - step) : step;

        // prefetch next step's Zx slice (128 KB): one 128B line per thread
        if (step + 1 < T_) {
            int tn = dir ? (T_ - 2 - step) : (step + 1);
            __builtin_prefetch(zdir + (size_t)tn * 32 * G4_ + tid * 32, 0, 1);
        }

        // init accumulators from Zx
        v8f acc[2][2];
        const float* zx = zdir + (size_t)t * 32 * G4_;
#pragma unroll
        for (int mt = 0; mt < 2; ++mt)
#pragma unroll
            for (int nl = 0; nl < 2; ++nl) {
                int n = (wave * 2 + nl) * 16 + q;
#pragma unroll
                for (int r = 0; r < 8; ++r)
                    acc[mt][nl][r] = zx[(size_t)(mt * 16 + r + hl * 8) * G4_ + n];
            }

        // recurrent GEMM: K = 256 in 8 WMMA steps
#pragma unroll
        for (int kt = 0; kt < 8; ++kt) {
            v16bf a[2];
#pragma unroll
            for (int mt = 0; mt < 2; ++mt) {
                int base = (mt * 16 + q) * HH_ + kt * 32 + hl * 8;
                v8bf lo = *(const v8bf*)(hbuf + base);
                v8bf hi = *(const v8bf*)(hbuf + base + 16);
                a[mt] = cat8(lo, hi);
            }
#pragma unroll
            for (int nl = 0; nl < 2; ++nl) {
                v16bf bw = wfrag[(size_t)kt * 2048 + (wave * 2 + nl) * 32 + lane];
#pragma unroll
                for (int mt = 0; mt < 2; ++mt)
                    acc[mt][nl] = __builtin_amdgcn_wmma_f32_16x16x32_bf16(
                        false, a[mt], false, bw, (short)0, acc[mt][nl], false, false);
            }
        }

        // publish z tiles to LDS
#pragma unroll
        for (int mt = 0; mt < 2; ++mt)
#pragma unroll
            for (int nl = 0; nl < 2; ++nl) {
                int n = (wave * 2 + nl) * 16 + q;
#pragma unroll
                for (int r = 0; r < 8; ++r)
                    zbuf[(mt * 16 + r + hl * 8) * G4_ + n] = (__bf16)acc[mt][nl][r];
            }
        __syncthreads();

        // gate nonlinearities; c in registers
        float hn[8];
#pragma unroll
        for (int u = 0; u < 8; ++u) {
            int j = ej + u;
            float zi = (float)zbuf[eb * G4_ + j];
            float zf = (float)zbuf[eb * G4_ + 256 + j];
            float zg = (float)zbuf[eb * G4_ + 512 + j];
            float zo = (float)zbuf[eb * G4_ + 768 + j];
            float cn = sigf(zf) * creg[u] + sigf(zi) * tanhf(zg);
            creg[u] = cn;
            hn[u] = sigf(zo) * tanhf(cn);
        }
#pragma unroll
        for (int u = 0; u < 8; ++u) {
            hbuf[eb * HH_ + ej + u] = (__bf16)hn[u];
            hs[(((size_t)dir * T_ + t) * B_ + eb) * HH_ + ej + u] = hn[u];
        }
        __syncthreads();
    }
}

// ---------------------------------------------------------------------------
// Output projection: feats_bt[b][t][tag] = concat(hf,hb)·W_out[tag] + b_out[tag]
// ---------------------------------------------------------------------------
__global__ __launch_bounds__(256) void out_proj(
    const float* __restrict__ hs, const float* __restrict__ Wout,
    const float* __restrict__ bout, float* __restrict__ feats) {
    int idx = blockIdx.x * 256 + threadIdx.x;
    if (idx >= B_ * T_ * NTG_) return;
    int tag = idx % NTG_;
    int t   = (idx / NTG_) % T_;
    int b   = idx / (NTG_ * T_);
    const float* hf = hs + ((size_t)t * B_ + b) * HH_;
    const float* hb = hs + (((size_t)T_ + t) * B_ + b) * HH_;
    const float* w  = Wout + tag * 512;
    float s = bout[tag];
#pragma unroll 4
    for (int k = 0; k < HH_; ++k) s += hf[k] * w[k];
#pragma unroll 4
    for (int k = 0; k < HH_; ++k) s += hb[k] * w[256 + k];
    feats[idx] = s;
}

// ---------------------------------------------------------------------------
// Viterbi decode per batch element: 1 block, 32 threads (1 per sequence).
// ---------------------------------------------------------------------------
__global__ __launch_bounds__(32) void viterbi(
    const float* __restrict__ feats, const float* __restrict__ trans,
    unsigned char* __restrict__ bp, float* __restrict__ out) {
    __shared__ float tr[NTG_ * NTG_];
    int tid = threadIdx.x;
    for (int i = tid; i < NTG_ * NTG_; i += 32) tr[i] = trans[i];
    __syncthreads();

    int b = tid;
    float fv[NTG_];
#pragma unroll
    for (int j = 0; j < NTG_; ++j) fv[j] = (j == 10) ? 0.0f : NEGV;

    const float* fb = feats + (size_t)b * T_ * NTG_;
    unsigned char* bpb = bp + (size_t)b * T_ * NTG_;

    for (int t = 0; t < T_; ++t) {
        float nf[NTG_];
#pragma unroll
        for (int nx = 0; nx < NTG_; ++nx) {
            float best = -3.4e38f; int arg = 0;
#pragma unroll
            for (int pv = 0; pv < NTG_; ++pv) {
                float v = fv[pv] + tr[nx * NTG_ + pv];
                if (v > best) { best = v; arg = pv; }
            }
            nf[nx] = best + fb[t * NTG_ + nx];
            bpb[t * NTG_ + nx] = (unsigned char)arg;
        }
#pragma unroll
        for (int j = 0; j < NTG_; ++j) fv[j] = nf[j];
    }

    float best = -3.4e38f; int bi = 0;
#pragma unroll
    for (int j = 0; j < NTG_; ++j) {
        float v = fv[j] + tr[11 * NTG_ + j];
        if (v > best) { best = v; bi = j; }
    }
    out[b] = best;

    int tag = bi;
    for (int t = T_ - 1; t >= 0; --t) {
        out[B_ + b * T_ + t] = (float)tag;
        tag = bpb[t * NTG_ + tag];
    }
}

// ---------------------------------------------------------------------------
extern "C" void kernel_launch(void* const* d_in, const int* in_sizes, int n_in,
                              void* d_out, int out_size, void* d_ws, size_t ws_size,
                              hipStream_t stream) {
    (void)in_sizes; (void)n_in; (void)out_size; (void)ws_size;

    const int*   sent  = (const int*)d_in[0];
    const float* embed = (const float*)d_in[1];
    const float* Wihf  = (const float*)d_in[2];
    const float* Whhf  = (const float*)d_in[3];
    const float* bf    = (const float*)d_in[4];
    const float* Wihb  = (const float*)d_in[5];
    const float* Whhb  = (const float*)d_in[6];
    const float* bb    = (const float*)d_in[7];
    const float* Wout  = (const float*)d_in[8];
    const float* bout  = (const float*)d_in[9];
    const float* trans = (const float*)d_in[10];
    const float* h0    = (const float*)d_in[11];
    const float* c0    = (const float*)d_in[12];

    char* w = (char*)d_ws;
    float*         Zx    = (float*)w;                       // 64 MB
    float*         hs    = (float*)(w + 67108864);          // 16 MB
    __bf16*        pack  = (__bf16*)(w + 83886080);         // 2 MB
    __bf16*        xsbf  = (__bf16*)(w + 85983232);         // 4 MB
    float*         feats = (float*)(w + 90177536);          // 384 KB
    unsigned char* bpbuf = (unsigned char*)(w + 90570752);  // 96 KB

    pack_weights<<<4096, 256, 0, stream>>>(Wihf, Wihb, Whhf, Whhb, pack);
    embed_pack<<<2048, 256, 0, stream>>>(sent, embed, xsbf);
    input_gemm<<<2048, 256, 0, stream>>>(xsbf, pack, bf, bb, Zx);
    lstm_rec<<<2, 1024, 0, stream>>>(Zx, pack, h0, c0, hs);
    out_proj<<<(B_ * T_ * NTG_ + 255) / 256, 256, 0, stream>>>(hs, Wout, bout, feats);
    viterbi<<<1, 32, 0, stream>>>(feats, trans, bpbuf, (float*)d_out);
}
